// MixtureOfExpertsLayer_86990267613688
// MI455X (gfx1250) — matrix operand order repaired and checked by reading, not compiled, not run on hardware
//
#include <hip/hip_runtime.h>
#include <hip/hip_bf16.h>
#include <math.h>

#define T_TOK   65536      // B*S
#define DIM     512
#define HID     1024
#define NEXP    8
#define MT      32         // token tile per block (LDS H-tile = 32*1024*2 = 64KB)

typedef __bf16  bf16_t;
typedef bf16_t  v16bf __attribute__((ext_vector_type(16)));
typedef float   v8f   __attribute__((ext_vector_type(8)));

struct __attribute__((aligned(16))) U4 { unsigned int a, b, c, d; };
union Frag { v16bf v; unsigned int u[8]; };

__device__ __forceinline__ unsigned short f2bf(float f) {
  unsigned int u = __float_as_uint(f);
  u += 0x7FFFu + ((u >> 16) & 1u);           // round-to-nearest-even
  return (unsigned short)(u >> 16);
}

__device__ __forceinline__ void load_frag(Frag& f, const unsigned short* p0,
                                          const unsigned short* p1) {
  U4 q0 = *(const U4*)p0;
  U4 q1 = *(const U4*)p1;
  f.u[0] = q0.a; f.u[1] = q0.b; f.u[2] = q0.c; f.u[3] = q0.d;
  f.u[4] = q1.a; f.u[5] = q1.b; f.u[6] = q1.c; f.u[7] = q1.d;
}

__device__ __forceinline__ float gelu_exact(float v) {
  return 0.5f * v * (1.0f + erff(v * 0.70710678118654752f));
}

// ---------------- init: zero output + control words ----------------
// ctl[0..7]=counts, ctl[8..15]=offsets, ctl[16..23]=cursors
__global__ void k_zero(float* out, int* ctl, long n) {
  long i  = (long)blockIdx.x * blockDim.x + threadIdx.x;
  long st = (long)gridDim.x * blockDim.x;
  for (; i < n; i += st) out[i] = 0.0f;
  if (blockIdx.x == 0 && threadIdx.x < 32) ctl[threadIdx.x] = 0;
}

// ---------------- convert activations f32 -> bf16 ----------------
__global__ void k_cvt_x(const float* x, unsigned short* xb, long n) {
  long i  = (long)blockIdx.x * blockDim.x + threadIdx.x;
  long st = (long)gridDim.x * blockDim.x;
  for (; i < n; i += st) xb[i] = f2bf(x[i]);
}

// ------------- convert + transpose expert weights to bf16 -------------
// w1t[e][h][d] = we1[e][d][h]   (N-major so WMMA B frags are contiguous)
// w2t[e][d][h] = we2[e][h][d]
__global__ void k_cvt_w(const float* we1, const float* we2,
                        unsigned short* w1t, unsigned short* w2t) {
  const long n  = (long)NEXP * DIM * HID;
  long i  = (long)blockIdx.x * blockDim.x + threadIdx.x;
  long st = (long)gridDim.x * blockDim.x;
  for (; i < 2 * n; i += st) {
    if (i < n) {
      long e = i / ((long)HID * DIM), r = i % ((long)HID * DIM);
      long h = r / DIM, d = r % DIM;
      w1t[i] = f2bf(we1[(e * DIM + d) * HID + h]);
    } else {
      long j = i - n;
      long e = j / ((long)DIM * HID), r = j % ((long)DIM * HID);
      long d = r / HID, h = r % HID;
      w2t[j] = f2bf(we2[(e * HID + h) * DIM + d]);
    }
  }
}

// ---------------- gating: one wave per token ----------------
__global__ __launch_bounds__(256) void k_gate(
    const float* __restrict__ x,  const float* __restrict__ gw1,
    const float* __restrict__ gb1, const float* __restrict__ gw2,
    const float* __restrict__ gb2,
    int* topIdx, float* topW, int* ctl) {
  __shared__ float sx[8][DIM];
  __shared__ float sh[8][256];
  __shared__ float slog[8][8];
  const int tid = threadIdx.x;
  const long tbase = (long)blockIdx.x * 8;
  for (int i = tid; i < 8 * DIM; i += 256)
    sx[i >> 9][i & 511] = x[(tbase + (i >> 9)) * DIM + (i & 511)];
  __syncthreads();
  const int wv = tid >> 5, ln = tid & 31;
  float acc[8];
#pragma unroll
  for (int j = 0; j < 8; j++) acc[j] = gb1[ln + 32 * j];
  for (int d = 0; d < DIM; d++) {
    const float xv = sx[wv][d];
    const float* g = &gw1[d * 256 + ln];
#pragma unroll
    for (int j = 0; j < 8; j++) acc[j] = fmaf(xv, g[32 * j], acc[j]);
  }
#pragma unroll
  for (int j = 0; j < 8; j++) sh[wv][ln + 32 * j] = fmaxf(acc[j], 0.0f);
  __syncthreads();
  if (ln < 8) {
    float l = gb2[ln];
    for (int h = 0; h < 256; h++) l = fmaf(sh[wv][h], gw2[h * 8 + ln], l);
    slog[wv][ln] = l;
  }
  __syncthreads();
  if (ln == 0) {
    float l[8], p[8], mx = -1e30f, s = 0.0f;
    for (int e = 0; e < 8; e++) { l[e] = slog[wv][e]; mx = fmaxf(mx, l[e]); }
    for (int e = 0; e < 8; e++) { p[e] = __expf(l[e] - mx); s += p[e]; }
    const float inv = 1.0f / s;
    for (int e = 0; e < 8; e++) p[e] *= inv;
    int i0 = 0;
    for (int e = 1; e < 8; e++) if (p[e] > p[i0]) i0 = e;
    int i1 = (i0 == 0) ? 1 : 0;
    for (int e = 0; e < 8; e++) if (e != i0 && p[e] > p[i1]) i1 = e;
    // re-softmax over the two selected probabilities (matches reference)
    const float e0 = __expf(p[i0] - p[i0]), e1 = __expf(p[i1] - p[i0]);
    const float ss = e0 + e1;
    const long t = tbase + wv;
    topIdx[2 * t] = i0;  topIdx[2 * t + 1] = i1;
    topW[2 * t] = e0 / ss; topW[2 * t + 1] = e1 / ss;
    atomicAdd(&ctl[i0], 1);
    atomicAdd(&ctl[i1], 1);
  }
}

// ---------------- exclusive scan of 8 counts ----------------
__global__ void k_scan(int* ctl) {
  if (threadIdx.x == 0) {
    int a = 0;
    for (int e = 0; e < NEXP; e++) { ctl[8 + e] = a; a += ctl[e]; }
  }
}

// ---------------- scatter tokens into per-expert lists ----------------
__global__ void k_scatter(const int* topIdx, const float* topW, int* ctl,
                          int* slotTok, float* slotW) {
  const int t = blockIdx.x * 256 + threadIdx.x;
  if (t >= T_TOK) return;
#pragma unroll
  for (int k = 0; k < 2; k++) {
    const int e = topIdx[2 * t + k];
    const int pos = atomicAdd(&ctl[16 + e], 1);
    const int s = ctl[8 + e] + pos;
    slotTok[s] = t;
    slotW[s]   = topW[2 * t + k];
  }
}

// ---------------- fused expert: (gather)GEMM1 -> GELU(LDS) -> GEMM2 -> scatter-add ----------------
__global__ __launch_bounds__(256) void k_expert(
    const unsigned short* __restrict__ xb,  const unsigned short* __restrict__ w1t,
    const unsigned short* __restrict__ w2t, const float* __restrict__ be1,
    const float* __restrict__ be2,          const int* __restrict__ ctl,
    const int* __restrict__ slotTok,        const float* __restrict__ slotW,
    float* out) {
  __shared__ unsigned short sH[MT * HID];   // 64 KB bf16 hidden tile

  // map flat block id -> (expert, tile-in-expert)
  int bid = blockIdx.x, e = -1, tile = 0, base = 0, cnt = 0, accb = 0;
  for (int i = 0; i < NEXP; i++) {
    const int c = ctl[i];
    const int nt = (c + MT - 1) / MT;
    if (bid < accb + nt) { e = i; tile = bid - accb; base = ctl[8 + i]; cnt = c; break; }
    accb += nt;
  }
  if (e < 0) return;
  const int m0 = tile * MT;
  const int tid = threadIdx.x, wv = tid >> 5;
  const int lane = tid & 31, ln = lane & 15, hi = lane >> 4;
  const v8f vzero = {0.f, 0.f, 0.f, 0.f, 0.f, 0.f, 0.f, 0.f};

  // ---------- GEMM1: H_tile = gelu(X_tile @ W1 + b1), bf16 -> LDS ----------
  const unsigned short* w1e = w1t + (size_t)e * HID * DIM;
  for (int mt = 0; mt < MT / 16; mt++) {
    const int srow = m0 + mt * 16 + ln;
    // out-of-range rows read token 0 (always valid); their contribution is
    // zeroed by the srow<cnt guard in the GEMM2 epilogue. No divergence.
    const int tok = (srow < cnt) ? slotTok[base + srow] : 0;
    const unsigned short* xrow = xb + (size_t)tok * DIM;
    v8f accv[8];
#pragma unroll
    for (int j = 0; j < 8; j++) accv[j] = vzero;
    // unroll 1: keep fragment live ranges within one iteration (no spills);
    // within the iteration, batch-issue A + all 8 B loads, then the WMMAs.
#pragma unroll 1
    for (int k0 = 0; k0 < DIM; k0 += 32) {
      Frag A;
      load_frag(A, xrow + k0 + hi * 8, xrow + k0 + hi * 8 + 16);
      Frag B[8];
#pragma unroll
      for (int j = 0; j < 8; j++) {
        const unsigned short* brow =
            w1e + (size_t)((wv * 8 + j) * 16 + ln) * DIM + k0 + hi * 8;
        load_frag(B[j], brow, brow + 16);
      }
#pragma unroll
      for (int j = 0; j < 8; j++)
        accv[j] = __builtin_amdgcn_wmma_f32_16x16x32_bf16(
            false, A.v, false, B[j].v, (short)0, accv[j], false, false);
    }
#pragma unroll
    for (int j = 0; j < 8; j++) {
      const int ncol = (wv * 8 + j) * 16 + ln;
      const float b1 = be1[(size_t)e * HID + ncol];
#pragma unroll
      for (int i = 0; i < 8; i++) {
        const int m = mt * 16 + i + hi * 8;
        sH[m * HID + ncol] = f2bf(gelu_exact(accv[j][i] + b1));
      }
    }
  }
  __syncthreads();

  // ---------- GEMM2: Y_tile = H_tile @ W2, weighted atomic scatter-add ----------
  const unsigned short* w2e = w2t + (size_t)e * DIM * HID;
  for (int mt = 0; mt < MT / 16; mt++) {
    v8f accv[4];
#pragma unroll
    for (int j = 0; j < 4; j++) accv[j] = vzero;
#pragma unroll 1
    for (int k0 = 0; k0 < HID; k0 += 32) {
      Frag A;
      const unsigned short* arow = &sH[(mt * 16 + ln) * HID + k0 + hi * 8];
      load_frag(A, arow, arow + 16);
      Frag B[4];
#pragma unroll
      for (int j = 0; j < 4; j++) {
        const unsigned short* brow =
            w2e + (size_t)((wv * 4 + j) * 16 + ln) * HID + k0 + hi * 8;
        load_frag(B[j], brow, brow + 16);
      }
#pragma unroll
      for (int j = 0; j < 4; j++)
        accv[j] = __builtin_amdgcn_wmma_f32_16x16x32_bf16(
            false, A.v, false, B[j].v, (short)0, accv[j], false, false);
    }
#pragma unroll
    for (int j = 0; j < 4; j++) {
      const int ncol = (wv * 4 + j) * 16 + ln;
      const float b2 = be2[(size_t)e * DIM + ncol];
#pragma unroll
      for (int i = 0; i < 8; i++) {
        const int m = mt * 16 + i + hi * 8;
        const int srow = m0 + m;
        if (srow < cnt) {
          const int tok = slotTok[base + srow];
          const float wgt = slotW[base + srow];
          atomicAdd(&out[(size_t)tok * DIM + ncol], wgt * (accv[j][i] + b2));
        }
      }
    }
  }
}

extern "C" void kernel_launch(void* const* d_in, const int* in_sizes, int n_in,
                              void* d_out, int out_size, void* d_ws, size_t ws_size,
                              hipStream_t stream) {
  const float* x   = (const float*)d_in[0];
  const float* gw1 = (const float*)d_in[1];
  const float* gb1 = (const float*)d_in[2];
  const float* gw2 = (const float*)d_in[3];
  const float* gb2 = (const float*)d_in[4];
  const float* we1 = (const float*)d_in[5];
  const float* be1 = (const float*)d_in[6];
  const float* we2 = (const float*)d_in[7];
  const float* be2 = (const float*)d_in[8];
  float* out = (float*)d_out;

  char* ws = (char*)d_ws;
  size_t off = 0;
  auto carve = [&](size_t bytes) -> char* {
    char* p = ws + off;
    off += (bytes + 255) & ~(size_t)255;
    return p;
  };
  unsigned short* xb  = (unsigned short*)carve((size_t)T_TOK * DIM * 2);   // 64 MB
  unsigned short* w1t = (unsigned short*)carve((size_t)NEXP * DIM * HID * 2);
  unsigned short* w2t = (unsigned short*)carve((size_t)NEXP * DIM * HID * 2);
  int*   topIdx  = (int*)carve((size_t)T_TOK * 2 * 4);
  float* topW    = (float*)carve((size_t)T_TOK * 2 * 4);
  int*   slotTok = (int*)carve((size_t)T_TOK * 2 * 4);
  float* slotW   = (float*)carve((size_t)T_TOK * 2 * 4);
  int*   ctl     = (int*)carve(32 * 4);

  k_zero   <<<2048, 256, 0, stream>>>(out, ctl, (long)T_TOK * DIM);
  k_cvt_x  <<<2048, 256, 0, stream>>>(x, xb, (long)T_TOK * DIM);
  k_cvt_w  <<<2048, 256, 0, stream>>>(we1, we2, w1t, w2t);
  k_gate   <<<T_TOK / 8, 256, 0, stream>>>(x, gw1, gb1, gw2, gb2, topIdx, topW, ctl);
  k_scan   <<<1, 32, 0, stream>>>(ctl);
  k_scatter<<<T_TOK / 256, 256, 0, stream>>>(topIdx, topW, ctl, slotTok, slotW);
  const int maxTiles = 2 * T_TOK / MT + NEXP;   // upper bound on per-expert tiles
  k_expert <<<maxTiles, 256, 0, stream>>>(xb, w1t, w2t, be1, be2, ctl,
                                          slotTok, slotW, out);
}
